// OTFlow_17927193493745
// MI455X (gfx1250) — matrix operand order repaired and checked by reading, not compiled, not run on hardware
//
#include <hip/hip_runtime.h>
#include <hip/hip_bf16.h>

typedef unsigned int u32;
typedef __attribute__((ext_vector_type(4)))  u32      v4u;
typedef __attribute__((ext_vector_type(8)))  u32      v8u;
typedef __attribute__((ext_vector_type(16))) _Float16 v16h;
typedef __attribute__((ext_vector_type(2)))  _Float16 h2;
typedef __attribute__((ext_vector_type(8)))  float    v8f;

#define DIM   64
#define DP1   65
#define MM    512
#define RR    10
#define PITCH 520   // Kopen^T row pitch in halves: 1040B = 16B aligned, bank-spread

union U16 { v16h h; v8u u; v4u q[2]; };

static __device__ __forceinline__ u32 pkmul(u32 a, u32 b) {
    union { u32 u; h2 h; } x, y, r;
    x.u = a; y.u = b;
    r.h = x.h * y.h;            // v_pk_mul_f16
    return r.u;
}

// ---------------------------------------------------------------------------
// Kernel 1a: symA = A^T A (65x65), rowSumSq(Kopen) (512), trace(symA[:64,:64])
// ---------------------------------------------------------------------------
__global__ __launch_bounds__(256) void otflow_prep(
    const float* __restrict__ A, const float* __restrict__ K0,
    float* __restrict__ symA, float* __restrict__ rssq, float* __restrict__ traceSum)
{
    int tid = threadIdx.x;
    for (int idx = tid; idx < DP1 * DP1; idx += 256) {
        int i = idx / DP1, j = idx % DP1;
        float s = 0.f;
        #pragma unroll
        for (int r = 0; r < RR; ++r) s = fmaf(A[r * DP1 + i], A[r * DP1 + j], s);
        symA[idx] = s;
    }
    for (int j = tid; j < MM; j += 256) {
        float s = 0.f;
        #pragma unroll 8
        for (int k = 0; k < DIM; ++k) { float v = K0[j * DP1 + k]; s = fmaf(v, v, s); }
        rssq[j] = s;
    }
    if (tid == 0) {
        float s = 0.f;
        for (int r = 0; r < RR; ++r)
            for (int d = 0; d < DIM; ++d) { float v = A[r * DP1 + d]; s = fmaf(v, v, s); }
        traceSum[0] = s;
    }
}

// ---------------------------------------------------------------------------
// Kernel 1b: K1 (f32) -> K1h (f16), row-major (A-operand feed, L2 resident)
// ---------------------------------------------------------------------------
__global__ __launch_bounds__(256) void otflow_cvt_k1(
    const float* __restrict__ K1, _Float16* __restrict__ K1h)
{
    int idx = blockIdx.x * 256 + threadIdx.x;
    if (idx < MM * MM) K1h[idx] = (_Float16)K1[idx];
}

// ---------------------------------------------------------------------------
// Kernel 2: per-sample f32 pipeline. 512 threads, 8 samples per block.
// Writes dz_dt / cost_L / cost_HJB to d_out, and (tanh_open f16,
// dt_lin1*w f32, trH_base f32) to workspace for the WMMA kernel.
// ---------------------------------------------------------------------------
__global__ __launch_bounds__(512) void otflow_point(
    const float* __restrict__ x, const float* __restrict__ tptr,
    const float* __restrict__ K0, const float* __restrict__ b0,
    const float* __restrict__ K1, const float* __restrict__ b1,
    const float* __restrict__ w,  const float* __restrict__ c_w,
    const float* __restrict__ symA, const float* __restrict__ rssq,
    _Float16* __restrict__ cbuf, float* __restrict__ abuf,
    float* __restrict__ trHbase, float* __restrict__ out, int N)
{
    extern __shared__ float smf[];
    float* s_s  = smf;                 // [8][66]
    float* s_u0 = s_s  + 8 * 66;       // [8][512]  (reused as tanh_open*z1)
    float* s_to = s_u0 + 8 * MM;       // [8][512]
    float* s_z1 = s_to + 8 * MM;       // [8][512]
    float* s_tw = s_z1 + 8 * MM;       // [8][512]  (tanh_lin1*w, reused as reduce buf)
    float* s_gr = s_tw + 8 * MM;       // [8][66]

    const int tid = threadIdx.x;
    const int n0  = blockIdx.x * 8;
    const float tval = tptr[0];

    for (int q = tid; q < 8 * DP1; q += 512) {
        int g = q / DP1, d = q % DP1;
        s_s[g * 66 + d] = (d < DIM) ? x[(size_t)(n0 + g) * DIM + d] : tval;
    }
    __syncthreads();

    const int j = tid;                 // 0..511
    // ---- opening = s @ K0^T + b0 ; tanh / antideriv_tanh
    {
        float acc[8];
        #pragma unroll
        for (int g = 0; g < 8; ++g) acc[g] = b0[j];
        const float* k0r = K0 + j * DP1;
        for (int d = 0; d < DP1; ++d) {
            float kv = k0r[d];
            #pragma unroll
            for (int g = 0; g < 8; ++g) acc[g] = fmaf(kv, s_s[g * 66 + d], acc[g]);
        }
        #pragma unroll
        for (int g = 0; g < 8; ++g) {
            float op = acc[g];
            float to = tanhf(op);
            float ax = fabsf(op);
            s_u0[g * MM + j] = ax + log1pf(expf(-2.f * ax));
            s_to[g * MM + j] = to;
            cbuf[(size_t)(n0 + g) * MM + j] = (_Float16)to;
        }
    }
    __syncthreads();

    // ---- lin1 = u0 @ K1^T + b1 ; tanh ; tw = tanh_lin1*w ; a = dt_lin1*w
    {
        float acc[8];
        #pragma unroll
        for (int g = 0; g < 8; ++g) acc[g] = b1[j];
        const float* k1r = K1 + (size_t)j * MM;
        for (int jj = 0; jj < MM; ++jj) {
            float kv = k1r[jj];
            #pragma unroll
            for (int g = 0; g < 8; ++g) acc[g] = fmaf(kv, s_u0[g * MM + jj], acc[g]);
        }
        float wi = w[j];
        #pragma unroll
        for (int g = 0; g < 8; ++g) {
            float tl = tanhf(acc[g]);
            s_tw[g * MM + j] = tl * wi;
            abuf[(size_t)(n0 + g) * MM + j] = (1.f - tl * tl) * wi;
        }
    }
    __syncthreads();

    // ---- z1 = w + (tanh_lin1*w) @ K1  (column access: coalesced)
    {
        float acc[8];
        #pragma unroll
        for (int g = 0; g < 8; ++g) acc[g] = 0.f;
        for (int i2 = 0; i2 < MM; ++i2) {
            float kv = K1[(size_t)i2 * MM + j];
            #pragma unroll
            for (int g = 0; g < 8; ++g) acc[g] = fmaf(kv, s_tw[g * MM + i2], acc[g]);
        }
        float wj = w[j];
        #pragma unroll
        for (int g = 0; g < 8; ++g) {
            float z1 = wj + acc[g];                 // H = 1
            s_z1[g * MM + j] = z1;
            s_u0[g * MM + j] = s_to[g * MM + j] * z1;   // toz (reuse)
        }
    }
    __syncthreads();

    // ---- trH_base partials: dt_open * z1 * rowSumSq
    {
        float rj = rssq[j];
        #pragma unroll
        for (int g = 0; g < 8; ++g) {
            float to = s_to[g * MM + j];
            s_tw[g * MM + j] = (1.f - to * to) * s_z1[g * MM + j] * rj;
        }
    }
    __syncthreads();
    for (int off = 256; off > 0; off >>= 1) {
        if (tid < off) {
            #pragma unroll
            for (int g = 0; g < 8; ++g)
                s_tw[g * MM + tid] += s_tw[g * MM + tid + off];
        }
        __syncthreads();
    }
    if (tid < 8) trHbase[n0 + tid] = s_tw[tid * MM];

    // ---- z0 / grad / outputs
    if (tid < 8 * DP1) {
        int g = tid / DP1, d = tid % DP1;
        float acc0 = 0.f;
        for (int j2 = 0; j2 < MM; ++j2)
            acc0 = fmaf(s_u0[g * MM + j2], K0[j2 * DP1 + d], acc0);
        float accA = 0.f;
        const float* sArow = symA + d * DP1;
        #pragma unroll 5
        for (int e = 0; e < DP1; ++e) accA = fmaf(sArow[e], s_s[g * 66 + e], accA);
        float grad = acc0 + accA + c_w[d];
        s_gr[g * 66 + d] = grad;
        if (d < DIM) out[(size_t)(n0 + g) * DIM + d] = -grad;   // dz_dt (ALPHA0=1)
    }
    __syncthreads();
    if (tid < 8) {
        int g = tid;
        float cl = 0.f;
        #pragma unroll 8
        for (int d = 0; d < DIM; ++d) { float v = s_gr[g * 66 + d]; cl = fmaf(v, v, cl); }
        cl *= 0.5f;
        float dphidt = s_gr[g * 66 + DIM];
        out[(size_t)N * DP1 + (n0 + g)] = cl;                    // cost_L_dt
        out[(size_t)N * 66  + (n0 + g)] = fabsf(-dphidt + cl);   // cost_HJB_dt
    }
}

// ---------------------------------------------------------------------------
// Kernel 3: trace-Hessian t1 term via f16 WMMA (f32 accumulate).
// 1 wave = 1 sample; 4 waves/block share Kopen^T (f16, padded) in LDS.
// KJ = K1 @ (c ⊙ Kopen): per sample 512x512x64 GEMM = 2048 wmma_f32_16x16x32_f16.
// 4 row-tiles x 4 col-tiles per accumulation group: 16 independent WMMAs per
// K-step, B fragments built once per K-step (2 v_pk_mul_f16 per WMMA).
// ---------------------------------------------------------------------------
__global__ __launch_bounds__(128) void otflow_trh_wmma(
    const _Float16* __restrict__ K1h, const float* __restrict__ K0,
    const _Float16* __restrict__ cbuf, const float* __restrict__ abuf,
    const float* __restrict__ trHbase, const float* __restrict__ traceSum,
    float* __restrict__ out_dlogp)
{
    extern __shared__ char smem[];
    _Float16* kT  = (_Float16*)smem;                              // [64][PITCH]
    _Float16* csm = (_Float16*)(smem + (size_t)DIM * PITCH * 2);  // [4][512]
    float*    asmb = (float*)  (smem + (size_t)DIM * PITCH * 2 + 4 * MM * 2); // [4][512]

    const int tid  = threadIdx.x;
    const int wv   = tid >> 5;
    const int lane = tid & 31;
    const int n    = blockIdx.x * 4 + wv;

    // stage Kopen^T (shared) and per-sample c (f16), a=dt_lin1*w (f32)
    for (int idx = tid; idx < DIM * MM; idx += 128) {
        int k = idx >> 9, jj = idx & (MM - 1);
        kT[k * PITCH + jj] = (_Float16)K0[jj * DP1 + k];
    }
    for (int jj = lane; jj < MM; jj += 32) {
        csm[wv * MM + jj]  = cbuf[(size_t)n * MM + jj];
        asmb[wv * MM + jj] = abuf[(size_t)n * MM + jj];
    }
    __syncthreads();

    const int hi = lane >> 4;     // lane half (A/B layout split)
    const int lm = lane & 15;
    float t1 = 0.f;

    for (int ig = 0; ig < 8; ++ig) {          // 8 groups of 64 output rows
        const int i0 = ig * 64;
        v8f acc[4][4];
        #pragma unroll
        for (int t = 0; t < 4; ++t)
            #pragma unroll
            for (int kt = 0; kt < 4; ++kt) acc[t][kt] = (v8f)(0.f);

        const u32* k1row[4];
        #pragma unroll
        for (int t = 0; t < 4; ++t)
            k1row[t] = (const u32*)(K1h + (size_t)(i0 + 16 * t + lm) * MM);

        if (ig < 7) // prefetch next row group's K1 rows (global_prefetch_b8)
            __builtin_prefetch(K1h + (size_t)(i0 + 64 + lm) * MM, 0, 1);

        for (int js = 0; js < 16; ++js) {
            const int j0 = js * 32;
            // A fragments: K1 rows, contiguous K layout (K 16*hi..+15)
            U16 a[4];
            const int d0 = (j0 >> 1) + 8 * hi;
            #pragma unroll
            for (int t = 0; t < 4; ++t) {
                a[t].q[0] = *(const v4u*)(k1row[t] + d0);
                a[t].q[1] = *(const v4u*)(k1row[t] + d0 + 4);
            }

            // broadcast c pairs for this K-slice (per lane-half)
            const int jb = j0 + 16 * hi;
            const u32* cp = (const u32*)(csm + wv * MM + jb);
            u32 cd[8];
            #pragma unroll
            for (int v = 0; v < 8; ++v) cd[v] = cp[v];

            // build all 4 B fragments first (independent regs, no WMMA WAR nops)
            U16 b[4];
            #pragma unroll
            for (int kt = 0; kt < 4; ++kt) {
                const int k = kt * 16 + lm;
                const u32* kp = (const u32*)(kT + k * PITCH + jb);
                #pragma unroll
                for (int v = 0; v < 8; ++v) b[kt].u[v] = pkmul(kp[v], cd[v]);
            }

            // 16 back-to-back WMMAs
            #pragma unroll
            for (int kt = 0; kt < 4; ++kt)
                #pragma unroll
                for (int t = 0; t < 4; ++t)
                    acc[t][kt] = __builtin_amdgcn_wmma_f32_16x16x32_f16(
                        false, a[t].h, false, b[kt].h, (short)0, acc[t][kt],
                        false, false);
        }

        // t1 += a_i * KJ[i,k]^2 over this 64-row strip (C layout: M = r + 8*hi)
        #pragma unroll
        for (int t = 0; t < 4; ++t)
            #pragma unroll
            for (int r = 0; r < 8; ++r) {
                float ai = asmb[wv * MM + i0 + 16 * t + r + 8 * hi];
                #pragma unroll
                for (int kt = 0; kt < 4; ++kt) {
                    float v = acc[t][kt][r];
                    t1 = fmaf(ai * v, v, t1);
                }
            }
    }

    // wave32 reduction
    #pragma unroll
    for (int off = 16; off > 0; off >>= 1) t1 += __shfl_xor(t1, off, 32);
    if (lane == 0) {
        float trH = trHbase[n] + t1 + traceSum[0];   // H = 1
        out_dlogp[n] = -trH;                         // dlogp_dt (ALPHA0=1)
    }
}

// ---------------------------------------------------------------------------
extern "C" void kernel_launch(void* const* d_in, const int* in_sizes, int n_in,
                              void* d_out, int out_size, void* d_ws, size_t ws_size,
                              hipStream_t stream) {
    const float* x   = (const float*)d_in[0];
    const float* t   = (const float*)d_in[1];
    const float* K0  = (const float*)d_in[2];
    const float* b0  = (const float*)d_in[3];
    const float* K1  = (const float*)d_in[4];
    const float* b1  = (const float*)d_in[5];
    const float* w   = (const float*)d_in[6];
    const float* A   = (const float*)d_in[7];
    const float* c_w = (const float*)d_in[8];
    const int N = in_sizes[0] / DIM;   // 4096

    char* ws = (char*)d_ws;
    size_t o = 0;
    _Float16* K1h  = (_Float16*)(ws + o); o += (size_t)MM * MM * 2;
    _Float16* cbuf = (_Float16*)(ws + o); o += (size_t)N * MM * 2;
    float* abuf    = (float*)(ws + o);    o += (size_t)N * MM * 4;
    float* trHb    = (float*)(ws + o);    o += (size_t)N * 4;
    float* symA    = (float*)(ws + o);    o += ((DP1 * DP1 * 4 + 255) & ~255);
    float* rssq    = (float*)(ws + o);    o += MM * 4;
    float* trace   = (float*)(ws + o);    o += 256;

    otflow_prep<<<1, 256, 0, stream>>>(A, K0, symA, rssq, trace);
    otflow_cvt_k1<<<(MM * MM + 255) / 256, 256, 0, stream>>>(K1, K1h);

    size_t smem2 = (size_t)(2 * 8 * 66 + 4 * 8 * MM) * sizeof(float);   // ~69.8 KB
    otflow_point<<<N / 8, 512, smem2, stream>>>(x, t, K0, b0, K1, b1, w, c_w,
                                                symA, rssq, cbuf, abuf, trHb,
                                                (float*)d_out, N);

    size_t smem3 = (size_t)DIM * PITCH * 2 + 4 * MM * 2 + 4 * MM * 4;   // ~78.8 KB
    otflow_trh_wmma<<<N / 4, 128, smem3, stream>>>(K1h, K0, cbuf, abuf, trHb, trace,
                                                   (float*)d_out + (size_t)N * DIM);
}
